// DilatedAttention_21440476742468
// MI455X (gfx1250) — matrix operand order repaired and checked by reading, not compiled, not run on hardware
//
#include <hip/hip_runtime.h>

// ---------------------------------------------------------------------------
// Dilated attention, fused single pass for MI455X (gfx1250, wave32, WMMA).
// B=4, H=16, T=8192, D=64, rates S in {8,16,32,64}, dil in {2,4,8,16}.
// One wave per (b, h, 64-row chunk). Memory-bound: Q/K/V read once, out
// written once (~512MB total => ~22us at 23.3 TB/s HBM).
// Matrix math: v_wmma_f32_16x16x32_bf16 (fp32 accumulate, fp32 softmax).
// Softmax reductions: DPP16 butterflies (no LDS traffic).
// V tile transposed in LDS -> PV fragments are 2x ds_load_b128 each.
// Output: staged through LDS -> coalesced global_store_b128.
// ---------------------------------------------------------------------------

typedef __attribute__((ext_vector_type(16))) __bf16 v16bf;
typedef __attribute__((ext_vector_type(8)))  __bf16 v8bf;
typedef __attribute__((ext_vector_type(8)))  float  v8f;

#define T_LEN     8192
#define D_DIM     64
#define VT_STRIDE 80    // key-axis stride of transposed V tile (64 + 16 zero)

// DPP16 butterfly controls (within each 16-lane row == our half-wave groups)
#define DPP_XOR1 0xB1   // quad_perm(1,0,3,2)   -> lane ^ 1
#define DPP_XOR2 0x4E   // quad_perm(2,3,0,1)   -> lane ^ 2
#define DPP_XOR4 0x141  // row_half_mirror      -> lane ^ 7 (ok after xor1/2)
#define DPP_XOR8 0x140  // row_mirror           -> lane ^ 15 (ok after above)

template <int CTRL>
__device__ __forceinline__ float dpp_xchg(float x) {
  int yi = __builtin_amdgcn_update_dpp(0, __builtin_bit_cast(int, x), CTRL,
                                       0xf, 0xf, true);
  return __builtin_bit_cast(float, yi);
}

__device__ __forceinline__ float red_max16(float m) {
  m = fmaxf(m, dpp_xchg<DPP_XOR1>(m));
  m = fmaxf(m, dpp_xchg<DPP_XOR2>(m));
  m = fmaxf(m, dpp_xchg<DPP_XOR4>(m));
  m = fmaxf(m, dpp_xchg<DPP_XOR8>(m));
  return m;
}

__device__ __forceinline__ float red_sum16(float s) {
  s += dpp_xchg<DPP_XOR1>(s);
  s += dpp_xchg<DPP_XOR2>(s);
  s += dpp_xchg<DPP_XOR4>(s);
  s += dpp_xchg<DPP_XOR8>(s);
  return s;
}

__device__ __forceinline__ v8f wmma_bf(v16bf a, v16bf b, v8f c) {
  // D = A(16x32 bf16) * B(32x16 bf16) + C(16x16 f32)
  return __builtin_amdgcn_wmma_f32_16x16x32_bf16(
      /*neg_a=*/false, a, /*neg_b=*/false, b,
      /*c_mod=*/(short)0, c, /*reuse_a=*/false, /*reuse_b=*/false);
}

#define CVT4(dst, base, f4)                         \
  (dst)[(base) + 0] = (__bf16)(f4).x;               \
  (dst)[(base) + 1] = (__bf16)(f4).y;               \
  (dst)[(base) + 2] = (__bf16)(f4).z;               \
  (dst)[(base) + 3] = (__bf16)(f4).w;

// A-matrix fragment (16x32 bf16) from row-major f32: this lane holds row
// (lane&15); element e -> K = half*16+e.
__device__ __forceinline__ v16bf load_afrag(const float* rowp, int half) {
  v16bf a;
  const float4* p0 = (const float4*)(rowp + half * 16);
  const float4* p1 = (const float4*)(rowp + half * 16 + 8);
  float4 x0 = p0[0], x1 = p0[1], y0 = p1[0], y1 = p1[1];
  CVT4(a, 0, x0); CVT4(a, 4, x1); CVT4(a, 8, y0); CVT4(a, 12, y1);
  return a;
}

// B-matrix fragment (32x16 bf16), column per lane, K contiguous:
// element i -> K = half*16 + i. Source: 16 contiguous f32.
__device__ __forceinline__ v16bf load_bfrag(const float* p) {
  v16bf b;
  const float4* q = (const float4*)p;
  float4 a0 = q[0], a1 = q[1], a2 = q[2], a3 = q[3];
  CVT4(b, 0, a0); CVT4(b, 4, a1); CVT4(b, 8, a2); CVT4(b, 12, a3);
  return b;
}

// probs A-fragment from LDS (row-major 16x64 bf16), row = lane&15.
// Both 8-element runs are 16B-aligned contiguous -> ds_load_b128 pairs.
__device__ __forceinline__ v16bf load_pfrag(const __bf16* P, int row, int kc,
                                            int half) {
  const v8bf* p = (const v8bf*)(P + row * 64 + kc * 32 + half * 16);
  v8bf lo = p[0];           // K = kc*32 + half*16 + 0..7
  v8bf hi = p[1];           // K = kc*32 + half*16 + 8..15
  v16bf a;
#pragma unroll
  for (int i = 0; i < 8; ++i) { a[i] = lo[i]; a[8 + i] = hi[i]; }
  return a;
}

// V B-fragment from transposed LDS tile: column `col` (d index), elements are
// 16 consecutive key rows kb..kb+15 -> two aligned ds_load_b128.
__device__ __forceinline__ v16bf load_vfrag(const __bf16* Vt, int kb, int col) {
  const v8bf* p = (const v8bf*)(Vt + col * VT_STRIDE + kb);
  v8bf lo = p[0], hi = p[1];
  v16bf b;
#pragma unroll
  for (int i = 0; i < 8; ++i) { b[i] = lo[i]; b[8 + i] = hi[i]; }
  return b;
}

// One rate's attention for a 16-row query group.
//   S      : segment size (mask granularity)
//   NT     : number of 16-key tiles
//   R1     : rate-1 row-validity (only half-wave half == h&1 outputs)
//   KSTART : block-relative first key row (compile-time)
template <int S, int NT, bool R1, int KSTART>
__device__ __forceinline__ void attend(const float* __restrict__ Kb,
                                       const __bf16* Vt, __bf16* Ps, int t0,
                                       int g, int h, const v16bf* qa,
                                       v8f* outt, int lane) {
  const int half = lane >> 4;
  const int n = lane & 15;
  constexpr int KC = (NT + 1) / 2;              // 32-key PV chunks

  // ---- scores: Q (16x64) x K^T, per 16-key tile, fp32 accum -------------
  v8f sc[NT];
#pragma unroll
  for (int kt = 0; kt < NT; ++kt) {
    const int k0 = KSTART + kt * 16;
    const float* krow = Kb + (size_t)(t0 + k0 + n) * D_DIM;
    v16bf kb0 = load_bfrag(krow + half * 16);        // d = 0..31 chunk
    v16bf kb1 = load_bfrag(krow + 32 + half * 16);   // d = 32..63 chunk
    v8f c;
#pragma unroll
    for (int i = 0; i < 8; ++i) c[i] = 0.0f;
    c = wmma_bf(qa[0], kb0, c);
    c = wmma_bf(qa[1], kb1, c);
    // scale + mask: same segment AND causal (block-relative indices)
#pragma unroll
    for (int v = 0; v < 8; ++v) {
      int q = g * 16 + v + half * 8;
      int k = k0 + n;
      bool ok = ((q / S) == (k / S)) && (q >= k);
      c[v] = ok ? c[v] * 0.125f : -1.0e30f;
    }
    sc[kt] = c;
  }

  // ---- softmax (per row; rows live per-half, DPP16 butterflies) ---------
  float rv = 1.0f;
  if (R1) rv = (half == (h & 1)) ? 1.0f : 0.0f;   // rate-1 row selection
  float pr[NT][8];
#pragma unroll
  for (int v = 0; v < 8; ++v) {
    float m = -1.0e30f;
#pragma unroll
    for (int kt = 0; kt < NT; ++kt) m = fmaxf(m, sc[kt][v]);
    m = red_max16(m);
    float s = 0.0f;
#pragma unroll
    for (int kt = 0; kt < NT; ++kt) {
      float e = __expf(sc[kt][v] - m);
      pr[kt][v] = e;
      s += e;
    }
    s = red_sum16(s);
    float inv = rv / s;
#pragma unroll
    for (int kt = 0; kt < NT; ++kt) pr[kt][v] *= inv;
  }

  // ---- probs C-layout -> LDS -> A-layout (zero-pad to 32-key multiple) --
  __syncthreads();                   // previous attend's LDS reads done
#pragma unroll
  for (int kt = 0; kt < NT; ++kt)
#pragma unroll
    for (int v = 0; v < 8; ++v)
      Ps[(v + half * 8) * 64 + kt * 16 + n] = (__bf16)pr[kt][v];
#pragma unroll
  for (int kt = NT; kt < KC * 2; ++kt)
#pragma unroll
    for (int v = 0; v < 8; ++v)
      Ps[(v + half * 8) * 64 + kt * 16 + n] = (__bf16)0.0f;
  __syncthreads();

  // ---- PV: probs (16 x 32k) x V (32k x 64), accumulate into out tiles ---
  // Keys past row 63 land in the zeroed pad rows of Vt (probs there are 0).
#pragma unroll
  for (int kc = 0; kc < KC; ++kc) {
    v16bf pa = load_pfrag(Ps, n, kc, half);
    int kb = KSTART + kc * 32 + half * 16;
#pragma unroll
    for (int dt = 0; dt < 4; ++dt) {
      v16bf vb = load_vfrag(Vt, kb, dt * 16 + n);
      outt[dt] = wmma_bf(pa, vb, outt[dt]);
    }
  }
}

// Process one 16-row query group G (0..3) of a 64-row chunk: all four rates.
template <int G>
__device__ __forceinline__ void process_group(const float* __restrict__ Qb,
                                              const float* __restrict__ Kb,
                                              const __bf16* Vt, __bf16* Ps,
                                              float* Os,
                                              float* __restrict__ Ob, int t0,
                                              int c, int h, int lane) {
  const int half = lane >> 4;
  const int n = lane & 15;

  // Q A-fragments for this group (d chunks 0..31 and 32..63), kept in VGPRs.
  const float* qrow = Qb + (size_t)(t0 + G * 16 + n) * D_DIM;
  v16bf qa[2];
  qa[0] = load_afrag(qrow, half);
  qa[1] = load_afrag(qrow + 32, half);

  v8f outt[4];
#pragma unroll
  for (int dt = 0; dt < 4; ++dt)
#pragma unroll
    for (int i = 0; i < 8; ++i) outt[dt][i] = 0.0f;

  // rate 1 (S=8, dil=2): always; selection handled per half-wave inside.
  attend<8, 1, true, G * 16>(Kb, Vt, Ps, t0, G, h, qa, outt, lane);
  // rate 2 (S=16, dil=4): this group is its own segment.
  if (G == (h & 3))
    attend<16, 1, false, G * 16>(Kb, Vt, Ps, t0, G, h, qa, outt, lane);
  // rate 3 (S=32, dil=8): 32-row segment containing this group.
  if ((((c << 1) + (G >> 1)) & 7) == (h & 7))
    attend<32, (G & 1) + 1, false, (G >> 1) * 32>(Kb, Vt, Ps, t0, G, h, qa,
                                                  outt, lane);
  // rate 4 (S=64, dil=16): whole chunk is one segment.
  if ((c & 15) == (h & 15))
    attend<64, G + 1, false, 0>(Kb, Vt, Ps, t0, G, h, qa, outt, lane);

  // ---- average over 4 rates; stage through LDS for b128 stores ----------
  __syncthreads();                  // previous group's Os reads done
#pragma unroll
  for (int dt = 0; dt < 4; ++dt)
#pragma unroll
    for (int v = 0; v < 8; ++v)
      Os[(v + half * 8) * 64 + dt * 16 + n] = outt[dt][v] * 0.25f;
  __syncthreads();
  // 16 rows x 64 f32 = 256 float4; 8 per lane, fully coalesced b128 stores.
#pragma unroll
  for (int it = 0; it < 8; ++it) {
    int fi = it * 32 + lane;        // float4 index 0..255
    int row = fi >> 4;              // 16 float4 per row
    int c4 = (fi & 15) << 2;
    float4 x = *(const float4*)&Os[row * 64 + c4];
    *(float4*)&Ob[(size_t)(t0 + G * 16 + row) * D_DIM + c4] = x;
  }
}

__global__ __launch_bounds__(32) void dilated_attn_kernel(
    const float* __restrict__ Q, const float* __restrict__ K,
    const float* __restrict__ V, float* __restrict__ Out) {
  __shared__ __bf16 Vt[64 * VT_STRIDE];  // V tile, TRANSPOSED: Vt[d][k]
  __shared__ __bf16 Ps[16 * 64];         // probs staging (C -> A layout)
  __shared__ float  Os[16 * 64];         // out staging for b128 stores

  const int lane = threadIdx.x;
  const int blk = blockIdx.x;        // 64 (b*h) * 128 chunks
  const int c = blk & 127;           // chunk index (T/64 = 128)
  const int bh = blk >> 7;
  const int h = bh & 15;
  const size_t base = (size_t)bh * (size_t)(T_LEN * D_DIM);
  const float* Qb = Q + base;
  const float* Kb = K + base;
  const float* Vb = V + base;
  float* Ob = Out + base;
  const int t0 = c << 6;

  // Warm Q/K rows of this chunk (global_prefetch) while V staging runs.
  __builtin_prefetch(Qb + (size_t)(t0 + lane) * D_DIM, 0, 3);
  __builtin_prefetch(Qb + (size_t)(t0 + 32 + lane) * D_DIM, 0, 3);
  __builtin_prefetch(Kb + (size_t)(t0 + lane) * D_DIM, 0, 3);
  __builtin_prefetch(Kb + (size_t)(t0 + 32 + lane) * D_DIM, 0, 3);

  // Zero the pad key-rows 64..79 of every column (reads there must be 0.0).
#pragma unroll
  for (int it = 0; it < 4; ++it) {
    int idx = it * 32 + lane;        // 0..127: (col, 8-row chunk)
    int col = idx >> 1;
    int ro = (idx & 1) << 3;
    v8bf z;
#pragma unroll
    for (int i = 0; i < 8; ++i) z[i] = (__bf16)0.0f;
    *(v8bf*)&Vt[col * VT_STRIDE + 64 + ro] = z;
  }

  // Stage V chunk (64x64 f32), b128 global loads, TRANSPOSE into LDS bf16.
#pragma unroll
  for (int it = 0; it < 32; ++it) {
    int fi = it * 32 + lane;          // float4 index 0..1023
    int row = fi >> 4;                // key row k
    int col4 = (fi & 15) << 2;        // d column base
    float4 x = ((const float4*)(Vb + (size_t)(t0 + row) * D_DIM))[fi & 15];
    Vt[(col4 + 0) * VT_STRIDE + row] = (__bf16)x.x;
    Vt[(col4 + 1) * VT_STRIDE + row] = (__bf16)x.y;
    Vt[(col4 + 2) * VT_STRIDE + row] = (__bf16)x.z;
    Vt[(col4 + 3) * VT_STRIDE + row] = (__bf16)x.w;
  }
  __syncthreads();

  process_group<0>(Qb, Kb, Vt, Ps, Os, Ob, t0, c, h, lane);
  process_group<1>(Qb, Kb, Vt, Ps, Os, Ob, t0, c, h, lane);
  process_group<2>(Qb, Kb, Vt, Ps, Os, Ob, t0, c, h, lane);
  process_group<3>(Qb, Kb, Vt, Ps, Os, Ob, t0, c, h, lane);
}

extern "C" void kernel_launch(void* const* d_in, const int* in_sizes, int n_in,
                              void* d_out, int out_size, void* d_ws,
                              size_t ws_size, hipStream_t stream) {
  (void)in_sizes; (void)n_in; (void)d_ws; (void)ws_size; (void)out_size;
  const float* Q = (const float*)d_in[0];
  const float* K = (const float*)d_in[1];
  const float* V = (const float*)d_in[2];
  float* Out = (float*)d_out;
  // B*H = 64 batch-heads, T/64 = 128 chunks -> 8192 one-wave workgroups.
  dim3 grid(64 * 128);
  dim3 block(32);
  dilated_attn_kernel<<<grid, block, 0, stream>>>(Q, K, V, Out);
}
